// GINNet_46617575031250
// MI455X (gfx1250) — compile-verified, matrix operand order checked by reading, not run on hardware
//
#include <hip/hip_runtime.h>
#include <stdint.h>

#define Hdim 128
#define Cdim 64

typedef __attribute__((ext_vector_type(16))) _Float16 v16h;
typedef __attribute__((ext_vector_type(8)))  _Float16 v8h;
typedef __attribute__((ext_vector_type(8)))  float    v8f;

// ---------------------------------------------------------------------------
// helpers
// ---------------------------------------------------------------------------
__device__ __forceinline__ v8f wmma16(v16h a, v16h b, v8f c) {
  // D = A(16x32 f16) * B(32x16 f16) + C(16x16 f32)
  return __builtin_amdgcn_wmma_f32_16x16x32_f16(
      /*neg_a=*/false, a, /*neg_b=*/false, b,
      /*c_mod=*/(short)0, c, /*reuse_a=*/false, /*reuse_b=*/false);
}

// A/B fragment: per-lane 16 halves = two 8-half (16B) LDS chunks at +0 and +16.
__device__ __forceinline__ v16h ld_frag(const _Float16* p) {
  v8h lo = *(const v8h*)p;
  v8h hi = *(const v8h*)(p + 16);
  return __builtin_shufflevector(lo, hi, 0,1,2,3,4,5,6,7,8,9,10,11,12,13,14,15);
}

// load 8 consecutive f32 from global, convert to f16, store 16B chunk to LDS
__device__ __forceinline__ void cvt8(const float* __restrict__ src,
                                     _Float16* dst, bool inb) {
  float4 a = make_float4(0.f, 0.f, 0.f, 0.f);
  float4 b = make_float4(0.f, 0.f, 0.f, 0.f);
  if (inb) {
    a = ((const float4*)src)[0];
    b = ((const float4*)src)[1];
  }
  v8h h;
  h[0] = (_Float16)a.x; h[1] = (_Float16)a.y; h[2] = (_Float16)a.z; h[3] = (_Float16)a.w;
  h[4] = (_Float16)b.x; h[5] = (_Float16)b.y; h[6] = (_Float16)b.z; h[7] = (_Float16)b.w;
  *(v8h*)dst = h;
}

// CDNA5 async copy: 16B global -> LDS directly (no VGPR round trip).
// Low 32 bits of the flat shared address ARE the LDS byte offset (ISA 10.2).
__device__ __forceinline__ void async_copy16(_Float16* lds_dst,
                                             const _Float16* gsrc) {
  const uint32_t lds = (uint32_t)(uintptr_t)lds_dst;
  const uint64_t ga  = (uint64_t)(uintptr_t)gsrc;
  asm volatile("global_load_async_to_lds_b128 %0, %1, off"
               :: "v"(lds), "v"(ga) : "memory");
}
__device__ __forceinline__ void wait_async0() {
  asm volatile("s_wait_asynccnt 0x0" ::: "memory");
}

// ---------------------------------------------------------------------------
// prep: transpose weights to f16 [n][k], fold BN into per-column scale/bias
// ---------------------------------------------------------------------------
__global__ __launch_bounds__(256) void prep_kernel(
    const float* __restrict__ conv_w,  // [6][128][128] (k-major)
    const float* __restrict__ conv_b,  // [6][128]
    const float* __restrict__ bn_g, const float* __restrict__ bn_b,
    const float* __restrict__ bn_m, const float* __restrict__ bn_v,
    const float* __restrict__ fc_w,    // [128][64]
    _Float16* __restrict__ wT,         // [6][128 n][128 k] f16
    _Float16* __restrict__ fcT,        // [64 n][128 k] f16
    float* __restrict__ scl,           // [6][128]
    float* __restrict__ bia)           // [6][128]
{
  const int idx = blockIdx.x * 256 + threadIdx.x;
  if (idx < 6 * Hdim * Hdim) {                 // conv weight transpose
    const int l = idx >> 14;
    const int r = idx & 16383;
    const int n = r >> 7, k = r & 127;
    wT[idx] = (_Float16)conv_w[(l << 14) + (k << 7) + n];
  } else if (idx < 6 * Hdim * Hdim + Cdim * Hdim) {   // fc weight transpose
    const int r = idx - 6 * Hdim * Hdim;
    const int n = r >> 7, k = r & 127;
    fcT[r] = (_Float16)fc_w[k * Cdim + n];
  } else if (idx < 6 * Hdim * Hdim + Cdim * Hdim + 6 * Hdim) {  // BN fold
    const int t = idx - (6 * Hdim * Hdim + Cdim * Hdim);        // [6][128] flat
    const float s = bn_g[t] * rsqrtf(bn_v[t] + 1e-5f);
    scl[t] = s;
    bia[t] = (conv_b[t] - bn_m[t]) * s + bn_b[t];
  }
}

// ---------------------------------------------------------------------------
// aggregation: agg = h (copy) then agg[dst] += h[src] over edges
// ---------------------------------------------------------------------------
__global__ __launch_bounds__(256) void copy4_kernel(
    const float4* __restrict__ s, float4* __restrict__ d, int n) {
  const int i = blockIdx.x * 256 + threadIdx.x;
  if (i < n) d[i] = s[i];
}

__global__ __launch_bounds__(256) void scatter_kernel(
    const float* __restrict__ h, const int* __restrict__ ei,
    float* __restrict__ agg, int E) {
  const unsigned idx = blockIdx.x * 256u + threadIdx.x;
  if (idx >= (unsigned)E * 32u) return;
  const int e = (int)(idx >> 5);
  const int f = (int)(idx & 31u) * 4;
  const int sidx = ei[e];
  const int didx = ei[E + e];
  const float4 v = *(const float4*)(h + (size_t)sidx * Hdim + f);
  float* p = agg + (size_t)didx * Hdim + f;
  atomicAdd(p + 0, v.x);
  atomicAdd(p + 1, v.y);
  atomicAdd(p + 2, v.z);
  atomicAdd(p + 3, v.w);
}

// ---------------------------------------------------------------------------
// fused GIN MLP: 3x (Linear 128x128 -> BN -> ReLU) on a 128-row tile.
// Activation tile resident in LDS (f16) across layers; weights staged with
// async global->LDS copies, double buffered to overlap with WMMA bursts.
// ---------------------------------------------------------------------------
constexpr int SIN_LD = 136;  // halves; 68 dwords -> bank-conflict-free frags
constexpr int SW_LD  = 40;   // halves; 20 dwords -> bank-conflict-free frags

__device__ __forceinline__ void stage_w_async128(const _Float16* __restrict__ W,
                                                 int k0, _Float16* sWbuf,
                                                 int tid) {
  #pragma unroll
  for (int c = 0; c < 2; ++c) {        // 512 chunks of 8 halves, 2 per thread
    const int id = tid + c * 256;
    const int n  = id >> 2;
    const int kb = (id & 3) * 8;
    async_copy16(sWbuf + n * SW_LD + kb, W + n * Hdim + k0 + kb);
  }
}

__global__ __launch_bounds__(256) void gin_mlp3_kernel(
    const float* __restrict__ Ain,     // [M,128] f32 (aggregated features)
    const _Float16* __restrict__ WT,   // [3][128 n][128 k] f16
    const float* __restrict__ scl,     // [3][128]
    const float* __restrict__ bia,     // [3][128]
    float* __restrict__ Out,           // [M,128] f32
    int M)
{
  __shared__ __align__(16) _Float16 sIn[128 * SIN_LD];     // 34816 B
  __shared__ __align__(16) _Float16 sW[2][128 * SW_LD];    // 2 x 10240 B

  const int tid  = threadIdx.x;
  const int wave = tid >> 5;
  const int lane = tid & 31;
  const int m0   = blockIdx.x * 128;

  // load input tile 128x128 f32 -> f16 (2 threads/row, 64 halves each)
  {
    const int r  = tid >> 1;
    const int cb = (tid & 1) * 64;
    const bool inb = (m0 + r) < M;
    const float* src = Ain + (size_t)(m0 + r) * Hdim + cb;
    _Float16* dst = sIn + r * SIN_LD + cb;
    #pragma unroll
    for (int c = 0; c < 8; ++c) cvt8(src + c * 8, dst + c * 8, inb);
  }
  __syncthreads();

  const int halfsel = (lane >> 4) * 8;  // K sub-block select per ISA layout
  const int lrow    = lane & 15;
  const int colbase = lane & 15;        // C/D: N = lane&15
  const int rowadd  = (lane >> 4) * 8;  // C/D: lanes 16-31 hold M+8

  for (int l = 0; l < 3; ++l) {
    v8f acc[8] = {};
    const _Float16* W = WT + l * (Hdim * Hdim);

    // prologue: async-stage k-slice 0 into buffer 0
    stage_w_async128(W, 0, sW[0], tid);
    wait_async0();
    __syncthreads();

    for (int ks = 0; ks < 4; ++ks) {
      const int cur = ks & 1;
      // prefetch next slice into the other buffer while we compute
      if (ks < 3) stage_w_async128(W, (ks + 1) * 32, sW[cur ^ 1], tid);

      const v16h af =
          ld_frag(sIn + (wave * 16 + lrow) * SIN_LD + ks * 32 + halfsel);
      #pragma unroll
      for (int j = 0; j < 8; ++j) {
        const v16h bf = ld_frag(sW[cur] + (j * 16 + lrow) * SW_LD + halfsel);
        acc[j] = wmma16(af, bf, acc[j]);
      }

      wait_async0();    // next slice landed in LDS
      __syncthreads();  // everyone done reading cur before it's overwritten
    }

    // epilogue: y = relu(acc * scale + bias); feed next layer via LDS
    #pragma unroll
    for (int j = 0; j < 8; ++j) {
      const int n = j * 16 + colbase;
      const float s = scl[l * Hdim + n];
      const float b = bia[l * Hdim + n];
      #pragma unroll
      for (int i = 0; i < 8; ++i) {
        const int r = wave * 16 + rowadd + i;
        const float v = fmaxf(acc[j][i] * s + b, 0.f);
        if (l < 2) {
          sIn[r * SIN_LD + n] = (_Float16)v;
        } else if (m0 + r < M) {
          Out[(size_t)(m0 + r) * Hdim + n] = v;
        }
      }
    }
    if (l < 2) __syncthreads();
  }
}

// ---------------------------------------------------------------------------
// final FC: [M,128] @ [128,64] + bias  (WMMA), logits -> d_out
// ---------------------------------------------------------------------------
__global__ __launch_bounds__(256) void fc_kernel(
    const float* __restrict__ Ain, const _Float16* __restrict__ WT,  // [64][128]
    const float* __restrict__ fcb, float* __restrict__ Out, int M)
{
  __shared__ __align__(16) _Float16 sIn[128 * SIN_LD];
  __shared__ __align__(16) _Float16 sW[2][Cdim * SW_LD];

  const int tid  = threadIdx.x;
  const int wave = tid >> 5;
  const int lane = tid & 31;
  const int m0   = blockIdx.x * 128;

  {
    const int r  = tid >> 1;
    const int cb = (tid & 1) * 64;
    const bool inb = (m0 + r) < M;
    const float* src = Ain + (size_t)(m0 + r) * Hdim + cb;
    _Float16* dst = sIn + r * SIN_LD + cb;
    #pragma unroll
    for (int c = 0; c < 8; ++c) cvt8(src + c * 8, dst + c * 8, inb);
  }

  // stage slice 0: 64n x 32k = 256 chunks, 1 per thread
  {
    const int n  = tid >> 2;
    const int kb = (tid & 3) * 8;
    async_copy16(sW[0] + n * SW_LD + kb, WT + n * Hdim + kb);
  }
  wait_async0();
  __syncthreads();

  const int halfsel = (lane >> 4) * 8;
  const int lrow    = lane & 15;
  const int colbase = lane & 15;
  const int rowadd  = (lane >> 4) * 8;

  v8f acc[4] = {};
  for (int ks = 0; ks < 4; ++ks) {
    const int cur = ks & 1;
    if (ks < 3) {
      const int n  = tid >> 2;
      const int kb = (tid & 3) * 8;
      async_copy16(sW[cur ^ 1] + n * SW_LD + kb,
                   WT + n * Hdim + (ks + 1) * 32 + kb);
    }

    const v16h af =
        ld_frag(sIn + (wave * 16 + lrow) * SIN_LD + ks * 32 + halfsel);
    #pragma unroll
    for (int j = 0; j < 4; ++j) {
      const v16h bf = ld_frag(sW[cur] + (j * 16 + lrow) * SW_LD + halfsel);
      acc[j] = wmma16(af, bf, acc[j]);
    }

    wait_async0();
    __syncthreads();
  }

  #pragma unroll
  for (int j = 0; j < 4; ++j) {
    const int n = j * 16 + colbase;
    const float b = fcb[n];
    #pragma unroll
    for (int i = 0; i < 8; ++i) {
      const int r = wave * 16 + rowadd + i;
      if (m0 + r < M) Out[(size_t)(m0 + r) * Cdim + n] = acc[j][i] + b;
    }
  }
}

// ---------------------------------------------------------------------------
// in-place log_softmax over 64 classes; one wave32 per row, 2 cols/lane
// ---------------------------------------------------------------------------
__global__ __launch_bounds__(256) void lsm_kernel(float* __restrict__ logits, int M) {
  const int row  = blockIdx.x * 8 + (threadIdx.x >> 5);
  const int lane = threadIdx.x & 31;
  if (row >= M) return;
  float* p = logits + (size_t)row * Cdim;
  const float a = p[lane];
  const float b = p[lane + 32];
  float mx = fmaxf(a, b);
  #pragma unroll
  for (int o = 16; o > 0; o >>= 1) mx = fmaxf(mx, __shfl_xor(mx, o, 32));
  float s = __expf(a - mx) + __expf(b - mx);
  #pragma unroll
  for (int o = 16; o > 0; o >>= 1) s += __shfl_xor(s, o, 32);
  const float lse = mx + __logf(s);
  p[lane]      = a - lse;
  p[lane + 32] = b - lse;
}

// ---------------------------------------------------------------------------
// orchestration
// ---------------------------------------------------------------------------
extern "C" void kernel_launch(void* const* d_in, const int* in_sizes, int n_in,
                              void* d_out, int out_size, void* d_ws, size_t ws_size,
                              hipStream_t stream) {
  const float* x      = (const float*)d_in[0];
  const int*   ei     = (const int*)d_in[1];
  // d_in[2] = edge_attr: ignored by GIN (matches reference)
  const float* conv_w = (const float*)d_in[3];
  const float* conv_b = (const float*)d_in[4];
  const float* bn_g   = (const float*)d_in[5];
  const float* bn_b   = (const float*)d_in[6];
  const float* bn_m   = (const float*)d_in[7];
  const float* bn_v   = (const float*)d_in[8];
  const float* fc_w   = (const float*)d_in[9];
  const float* fc_b   = (const float*)d_in[10];

  const int N = in_sizes[0] / Hdim;
  const int E = in_sizes[1] / 2;

  // workspace layout (all offsets 16B-aligned)
  float*    hA  = (float*)d_ws;                       // [N,128] f32
  float*    hB  = hA + (size_t)N * Hdim;              // [N,128] f32
  _Float16* wT  = (_Float16*)(hB + (size_t)N * Hdim); // [6][128][128] f16
  _Float16* fcT = wT + 6 * Hdim * Hdim;               // [64][128] f16
  float*    scl = (float*)(fcT + Cdim * Hdim);        // [6][128]
  float*    bia = scl + 6 * Hdim;                     // [6][128]

  const int prep_threads = 6 * Hdim * Hdim + Cdim * Hdim + 6 * Hdim;
  prep_kernel<<<(prep_threads + 255) / 256, 256, 0, stream>>>(
      conv_w, conv_b, bn_g, bn_b, bn_m, bn_v, fc_w, wT, fcT, scl, bia);

  const int nv4       = N * (Hdim / 4);
  const int copy_grid = (nv4 + 255) / 256;
  const int gemm_grid = (N + 127) / 128;
  const int sc_grid   = (int)(((unsigned)E * 32u + 255u) / 256u);

  // ---- GIN conv 1: agg = x + scatter(x); 3x fused MLP ----
  copy4_kernel<<<copy_grid, 256, 0, stream>>>((const float4*)x, (float4*)hA, nv4);
  scatter_kernel<<<sc_grid, 256, 0, stream>>>(x, ei, hA, E);
  gin_mlp3_kernel<<<gemm_grid, 256, 0, stream>>>(hA, wT, scl, bia, hB, N);

  // ---- GIN conv 2 ----
  copy4_kernel<<<copy_grid, 256, 0, stream>>>((const float4*)hB, (float4*)hA, nv4);
  scatter_kernel<<<sc_grid, 256, 0, stream>>>(hB, ei, hA, E);
  gin_mlp3_kernel<<<gemm_grid, 256, 0, stream>>>(
      hA, wT + 3 * Hdim * Hdim, scl + 3 * Hdim, bia + 3 * Hdim, hB, N);

  // ---- FC + log_softmax (in place on d_out) ----
  fc_kernel<<<gemm_grid, 256, 0, stream>>>(hB, fcT, fc_b, (float*)d_out, N);
  lsm_kernel<<<(N + 7) / 8, 256, 0, stream>>>((float*)d_out, N);
}
